// NeighborhoodAttention1d_88553635708963
// MI455X (gfx1250) — matrix operand order repaired and checked by reading, not compiled, not run on hardware
//
#include <hip/hip_runtime.h>

// ---------------------------------------------------------------------------
// NeighborhoodAttention1d for MI455X (gfx1250, wave32, WMMA)
//   B=2, L=16384, C=512, H=16, D=32, K=7
// Pipeline:
//   1) fp32 -> bf16 conversion of x / qkv_w / proj_w        (memory-bound)
//   2) QKV GEMM  32768x1536x512 via v_wmma_f32_16x16x32_bf16
//   3) neighborhood attention (fp32 math, bf16 I/O)
//   4) proj GEMM 32768x512x512 via v_wmma_f32_16x16x32_bf16 -> fp32 out
// ---------------------------------------------------------------------------

typedef __attribute__((ext_vector_type(8)))  float          v8f;
typedef __attribute__((ext_vector_type(16))) __bf16         v16bf;
typedef __attribute__((ext_vector_type(8)))  unsigned short v8u;

struct Halves { v8u lo; v8u hi; };
union BFrag {
    Halves h;
    v16bf  v;
};

__device__ __forceinline__ float bf2f(unsigned short u) {
    unsigned int v = ((unsigned int)u) << 16;
    return __builtin_bit_cast(float, v);
}

__device__ __forceinline__ unsigned short f2bf(float f) {
    unsigned int x = __builtin_bit_cast(unsigned int, f);
    unsigned int r = x + 0x7FFFu + ((x >> 16) & 1u);   // round-to-nearest-even
    return (unsigned short)(r >> 16);
}

// ---------------------------------------------------------------------------
// fp32 -> bf16 elementwise conversion (grid-stride)
// ---------------------------------------------------------------------------
__global__ __launch_bounds__(256)
void f32_to_bf16_kernel(const float* __restrict__ in,
                        unsigned short* __restrict__ out,
                        size_t n)
{
    size_t i      = (size_t)blockIdx.x * blockDim.x + threadIdx.x;
    size_t stride = (size_t)gridDim.x * blockDim.x;
    for (; i < n; i += stride) out[i] = f2bf(in[i]);
}

// ---------------------------------------------------------------------------
// GEMM:  C[m,n] = sum_k A[m,k] * Bm[n,k] + bias[n]
//   A  : [M x Kd] bf16 row-major
//   Bm : [N x Kd] bf16 row-major   (i.e. computes A @ Bm^T)
//   C  : bf16 [M x N]  (OUT_BF16) or fp32 [M x N]
// One wave32 per 32x32 output tile; 2x2 grid of 16x16x32 WMMA tiles.
//
// VGPR fragment layouts (CDNA5 ISA 7.12.2, wave32):
//   A 16x32 bf16 : lane row = lane&15; per-lane 16 elems =
//                  k in [k0 + (lane>>4)*8, +8) ++ [k0 + (lane>>4)*8 + 16, +8)
//   B 32x16 bf16 : lane col = lane&15; per-lane 16 elems =
//                  k in [k0 + (lane>>4)*8, +8) ++ [k0 + (lane>>4)*8 + 16, +8)
//   C/D 16x16 f32: VGPR r -> row r + (lane>>4)*8, col = lane&15
// ---------------------------------------------------------------------------
template <bool OUT_BF16>
__global__ __launch_bounds__(256)
void gemm_bf16_wmma_kernel(const unsigned short* __restrict__ A,
                           const unsigned short* __restrict__ Bm,
                           const float* __restrict__ bias,
                           void* __restrict__ Cptr,
                           int M, int N, int Kd)
{
    const int lane = threadIdx.x & 31;
    const int wave = threadIdx.x >> 5;
    const int l16  = lane & 15;
    const int half = lane >> 4;

    const int tilesN = N >> 5;                       // 32-wide tiles in N
    const int tile   = blockIdx.x * (blockDim.x >> 5) + wave;
    const int tm     = (tile / tilesN) << 5;
    const int tn     = (tile % tilesN) << 5;
    if (tm >= M) return;

    const unsigned short* a0p = A  + (size_t)(tm + l16) * Kd;
    const unsigned short* a1p = a0p + (size_t)16 * Kd;
    const unsigned short* b0p = Bm + (size_t)(tn + l16) * Kd;
    const unsigned short* b1p = b0p + (size_t)16 * Kd;

    const int ca = half << 3;    // A per-lane k sub-offset (0 or 8)
    const int cb = half << 3;    // B per-lane k sub-offset (0 or 8)

    v8f acc00 = {}, acc01 = {}, acc10 = {}, acc11 = {};

    for (int k = 0; k < Kd; k += 32) {
        BFrag a0, a1, b0, b1;
        a0.h.lo = *(const v8u*)(a0p + k + ca);
        a0.h.hi = *(const v8u*)(a0p + k + ca + 16);
        a1.h.lo = *(const v8u*)(a1p + k + ca);
        a1.h.hi = *(const v8u*)(a1p + k + ca + 16);
        b0.h.lo = *(const v8u*)(b0p + k + cb);
        b0.h.hi = *(const v8u*)(b0p + k + cb + 16);
        b1.h.lo = *(const v8u*)(b1p + k + cb);
        b1.h.hi = *(const v8u*)(b1p + k + cb + 16);

        // hint-prefetch the next K-slab of the streaming A operand
        if (k + 32 < Kd) {
            __builtin_prefetch(a0p + k + 32 + ca, 0, 1);
            __builtin_prefetch(a1p + k + 32 + ca, 0, 1);
        }

        acc00 = __builtin_amdgcn_wmma_f32_16x16x32_bf16(
            false, a0.v, false, b0.v, (short)0, acc00, false, false);
        acc01 = __builtin_amdgcn_wmma_f32_16x16x32_bf16(
            false, a0.v, false, b1.v, (short)0, acc01, false, false);
        acc10 = __builtin_amdgcn_wmma_f32_16x16x32_bf16(
            false, a1.v, false, b0.v, (short)0, acc10, false, false);
        acc11 = __builtin_amdgcn_wmma_f32_16x16x32_bf16(
            false, a1.v, false, b1.v, (short)0, acc11, false, false);
    }

    const int   n0    = tn + l16;
    const int   n1    = n0 + 16;
    const float bias0 = bias[n0];
    const float bias1 = bias[n1];
    const int   rbase = half << 3;                   // 0 or 8

#pragma unroll
    for (int r = 0; r < 8; ++r) {
        const int m0 = tm + rbase + r;
        const int m1 = m0 + 16;
        const float v00 = acc00[r] + bias0;
        const float v01 = acc01[r] + bias1;
        const float v10 = acc10[r] + bias0;
        const float v11 = acc11[r] + bias1;
        if constexpr (OUT_BF16) {
            unsigned short* C = (unsigned short*)Cptr;
            C[(size_t)m0 * N + n0] = f2bf(v00);
            C[(size_t)m0 * N + n1] = f2bf(v01);
            C[(size_t)m1 * N + n0] = f2bf(v10);
            C[(size_t)m1 * N + n1] = f2bf(v11);
        } else {
            float* C = (float*)Cptr;
            C[(size_t)m0 * N + n0] = v00;
            C[(size_t)m0 * N + n1] = v01;
            C[(size_t)m1 * N + n0] = v10;
            C[(size_t)m1 * N + n1] = v11;
        }
    }
}

// ---------------------------------------------------------------------------
// Neighborhood attention, one thread per (token, head).
//   qkv : [B*L x 1536] bf16, row = [q(512) | k(512) | v(512)], head h at h*32
//   out : [B*L x 512]  bf16
// Lane -> head mapping keeps adjacent lanes on adjacent 64B chunks.
// ---------------------------------------------------------------------------
__global__ __launch_bounds__(256)
void natt1d_kernel(const unsigned short* __restrict__ qkv,
                   const float* __restrict__ rpb,
                   unsigned short* __restrict__ outp,
                   int BL, int L)
{
    constexpr int Hh = 16, Dd = 32, Kk = 7;
    constexpr float SCALE = 0.17677669529663687f;    // 32^-0.5

    const int gid = blockIdx.x * blockDim.x + threadIdx.x;
    const int h   = gid & (Hh - 1);
    const int m   = gid >> 4;                        // token index in [0, B*L)
    if (m >= BL) return;
    const int l = m & (L - 1);                       // position within sequence

    const unsigned short* qp = qkv + (size_t)m * 1536 + h * Dd;
    float q[Dd];
#pragma unroll
    for (int d = 0; d < Dd; ++d) q[d] = bf2f(qp[d]) * SCALE;

    int ni = l - (Kk / 2);
    if (ni < 0) ni = 0;
    if (ni > L - Kk) ni = L - Kk;

    const size_t batchBase = (size_t)(m - l) * 1536;

    float logit[Kk];
#pragma unroll
    for (int j = 0; j < Kk; ++j) {
        const unsigned short* kp =
            qkv + batchBase + (size_t)(ni + j) * 1536 + 512 + h * Dd;
        float s = 0.0f;
#pragma unroll
        for (int d = 0; d < Dd; ++d) s += q[d] * bf2f(kp[d]);
        logit[j] = s + rpb[h * (2 * Kk - 1) + (ni + j - l + (Kk - 1))];
    }

    float mx = logit[0];
#pragma unroll
    for (int j = 1; j < Kk; ++j) mx = fmaxf(mx, logit[j]);

    float p[Kk];
    float sum = 0.0f;
#pragma unroll
    for (int j = 0; j < Kk; ++j) {
        p[j] = __expf(logit[j] - mx);
        sum += p[j];
    }
    const float inv = 1.0f / sum;

    float o[Dd] = {};
#pragma unroll
    for (int j = 0; j < Kk; ++j) {
        const unsigned short* vp =
            qkv + batchBase + (size_t)(ni + j) * 1536 + 1024 + h * Dd;
        const float w = p[j] * inv;
#pragma unroll
        for (int d = 0; d < Dd; ++d) o[d] += w * bf2f(vp[d]);
    }

    unsigned short* op = outp + (size_t)m * 512 + h * Dd;
#pragma unroll
    for (int d = 0; d < Dd; ++d) op[d] = f2bf(o[d]);
}

// ---------------------------------------------------------------------------
// Host-side launcher
// ---------------------------------------------------------------------------
extern "C" void kernel_launch(void* const* d_in, const int* in_sizes, int n_in,
                              void* d_out, int out_size, void* d_ws, size_t ws_size,
                              hipStream_t stream)
{
    (void)in_sizes; (void)n_in; (void)out_size; (void)ws_size;

    constexpr int Bc = 2, Lc = 16384, Cc = 512;
    constexpr int Mc = Bc * Lc;          // 32768 tokens
    constexpr int N1 = 3 * Cc;           // 1536

    const float* x      = (const float*)d_in[0];
    const float* qkv_w  = (const float*)d_in[1];
    const float* qkv_b  = (const float*)d_in[2];
    const float* rpb    = (const float*)d_in[3];
    const float* proj_w = (const float*)d_in[4];
    const float* proj_b = (const float*)d_in[5];
    float* out          = (float*)d_out;

    // workspace layout (bf16 / ushort elements)
    const size_t n_x   = (size_t)Mc * Cc;    // 16.7M
    const size_t n_w1  = (size_t)N1 * Cc;    // 0.79M
    const size_t n_w2  = (size_t)Cc * Cc;    // 0.26M
    const size_t n_qkv = (size_t)Mc * N1;    // 50.3M

    unsigned short* x_bf    = (unsigned short*)d_ws;
    unsigned short* w1_bf   = x_bf + n_x;
    unsigned short* w2_bf   = w1_bf + n_w1;
    unsigned short* qkv_bf  = w2_bf + n_w2;
    unsigned short* attn_bf = qkv_bf + n_qkv;

    // 1) fp32 -> bf16 conversions
    f32_to_bf16_kernel<<<2048, 256, 0, stream>>>(x, x_bf, n_x);
    f32_to_bf16_kernel<<<512,  256, 0, stream>>>(qkv_w, w1_bf, n_w1);
    f32_to_bf16_kernel<<<256,  256, 0, stream>>>(proj_w, w2_bf, n_w2);

    // 2) QKV GEMM: [Mc x Cc] @ [N1 x Cc]^T -> bf16 [Mc x N1]
    {
        const int tiles  = (Mc / 32) * (N1 / 32);    // 49152
        const int blocks = tiles / 8;                // 8 waves per block
        gemm_bf16_wmma_kernel<true><<<blocks, 256, 0, stream>>>(
            x_bf, w1_bf, qkv_b, (void*)qkv_bf, Mc, N1, Cc);
    }

    // 3) neighborhood attention -> bf16 [Mc x Cc]
    {
        const int threads = Mc * 16;                 // one per (token, head)
        natt1d_kernel<<<threads / 256, 256, 0, stream>>>(
            qkv_bf, rpb, attn_bf, Mc, Lc);
    }

    // 4) proj GEMM: [Mc x Cc] @ [Cc x Cc]^T -> fp32 d_out
    {
        const int tiles  = (Mc / 32) * (Cc / 32);    // 16384
        const int blocks = tiles / 8;
        gemm_bf16_wmma_kernel<false><<<blocks, 256, 0, stream>>>(
            attn_bf, w2_bf, proj_b, (void*)out, Mc, Cc, Cc);
    }
}
